// AttentionCritic_35218731827276
// MI455X (gfx1250) — matrix operand order repaired and checked by reading, not compiled, not run on hardware
//
#include <hip/hip_runtime.h>

typedef __attribute__((ext_vector_type(16))) _Float16 v16h;
typedef __attribute__((ext_vector_type(8)))  _Float16 v8h;
typedef __attribute__((ext_vector_type(4)))  _Float16 v4h;
typedef __attribute__((ext_vector_type(8)))  float    v8f;

#define AGENTS 8
#define BATCH  32768
#define SDIM   256
#define ADIM   32
#define IDIM   288
#define HID    128
#define DHEAD  32
#define EPSBN  1e-3f
#define ALPHA  0.3f

#define BT     64      // batch rows per workgroup (GEMM kernels)
#define SA_STR 296     // padded LDS stride for [BT,288] f16 tile
#define E_STR  136     // padded LDS stride for [BT,128] f16 tile (16B-aligned rows)
#define C_STR  264     // padded LDS stride for [BT,256] f16 tile (16B-aligned rows)
#define TB     8       // batch rows per attention workgroup
#define AT_STR 136     // padded LDS stride for attention tiles (16B-aligned rows)

// packed-weight fragment counts (f16 elements)
#define WENC_N ((size_t)AGENTS*9*8*512)   // 294912
#define WS_N   ((size_t)AGENTS*8*8*512)   // 262144
#define WP_N   ((size_t)4*8*512)          // 16384 (per projection matrix)
#define WC1_N  ((size_t)AGENTS*8*8*512)   // 262144

static __device__ __forceinline__ v16h cat16(v8h lo, v8h hi) {
  return __builtin_shufflevector(lo, hi, 0,1,2,3,4,5,6,7,8,9,10,11,12,13,14,15);
}

// CDNA5 async copy: global -> LDS, 16 bytes per lane, GVS addressing
// (uniform SGPR base + per-lane 32-bit byte offset). Tracked by ASYNCcnt.
static __device__ __forceinline__ void async_cp16(unsigned lds_byte_off,
                                                  const void* uniform_base,
                                                  unsigned vbyte_off) {
  asm volatile("global_load_async_to_lds_b128 %0, %1, %2"
               :: "v"(lds_byte_off), "v"(vbyte_off),
                  "s"((unsigned long long)(size_t)uniform_base)
               : "memory");
}
static __device__ __forceinline__ void wait_async0() {
#if defined(__has_builtin)
#if __has_builtin(__builtin_amdgcn_s_wait_asynccnt)
  __builtin_amdgcn_s_wait_asynccnt(0);
#else
  asm volatile("s_wait_asynccnt 0" ::: "memory");
#endif
#else
  asm volatile("s_wait_asynccnt 0" ::: "memory");
#endif
}
static __device__ __forceinline__ unsigned lds_off(const void* p) {
  return (unsigned)(size_t)p;   // generic-from-LDS: low 32 bits == LDS byte address
}

// A (LDS, row-major, padded stride) x B (packed frags in ws) -> 16x128 f32 accum.
// Software-pipelined: prefetch the whole next k-step (A-frag + 8 B-frags) while
// the 8 WMMAs of the current k-step execute, so loads overlap the matrix pipe.
template<int KT>
static __device__ __forceinline__ void gemm_tile(const _Float16* A0, int astr,
                                                 const _Float16* B0, v8f acc[8],
                                                 int lane) {
  const int lh = lane >> 4, lm = lane & 15;
  const _Float16* arow = A0 + lm*astr + lh*16;
  const _Float16* bb   = B0 + (size_t)lane*16;
  v16h a, b[8];
  a = cat16(*(const v8h*)arow, *(const v8h*)(arow + 8));
  #pragma unroll
  for (int nt = 0; nt < 8; ++nt)
    b[nt] = cat16(*(const v8h*)(bb + nt*512), *(const v8h*)(bb + nt*512 + 8));
  #pragma unroll
  for (int kt = 0; kt < KT; ++kt) {
    v16h a2 = a, b2[8];
    #pragma unroll
    for (int nt = 0; nt < 8; ++nt) b2[nt] = b[nt];
    if (kt + 1 < KT) {               // issue next k-step loads before the WMMAs
      const _Float16* an = arow + (kt+1)*32;
      a = cat16(*(const v8h*)an, *(const v8h*)(an + 8));
      const _Float16* bn = bb + (size_t)(kt+1)*8*512;
      #pragma unroll
      for (int nt = 0; nt < 8; ++nt)
        b[nt] = cat16(*(const v8h*)(bn + nt*512), *(const v8h*)(bn + nt*512 + 8));
    }
    #pragma unroll
    for (int nt = 0; nt < 8; ++nt)
      acc[nt] = __builtin_amdgcn_wmma_f32_16x16x32_f16(false, a2, false, b2[nt],
                                                       (short)0, acc[nt], false, false);
  }
}

// ---------------- weight packing: BN fold + WMMA B-fragment layout ----------------
__global__ void k_pack_frags(const float* __restrict__ W_enc, const float* __restrict__ var_sa,
                             const float* __restrict__ W_s,   const float* __restrict__ var_s,
                             const float* __restrict__ W_k,   const float* __restrict__ W_v,
                             const float* __restrict__ W_sel, const float* __restrict__ W_c1,
                             _Float16* __restrict__ WencF, _Float16* __restrict__ WsF,
                             _Float16* __restrict__ WkF,   _Float16* __restrict__ WvF,
                             _Float16* __restrict__ WselF, _Float16* __restrict__ Wc1F) {
  size_t idx = (size_t)blockIdx.x*blockDim.x + threadIdx.x;
  if (idx < WENC_N) {                                  // W_enc, BN(sa) folded
    size_t a = idx / 36864, r = idx % 36864;
    int f = (int)(r/512), L = (int)((r%512)/16), e = (int)(r%16);
    int kt = f>>3, nt = f&7;
    int k = kt*32 + (L>>4)*16 + e, n = nt*16 + (L&15);
    float sc = rsqrtf(var_sa[a*IDIM + k] + EPSBN);
    WencF[idx] = (_Float16)(W_enc[(a*IDIM + k)*HID + n] * sc);
    return;
  }
  idx -= WENC_N;
  if (idx < WS_N) {                                    // W_s, BN(s) folded
    size_t a = idx / 32768, r = idx % 32768;
    int f = (int)(r/512), L = (int)((r%512)/16), e = (int)(r%16);
    int kt = f>>3, nt = f&7;
    int k = kt*32 + (L>>4)*16 + e, n = nt*16 + (L&15);
    float sc = rsqrtf(var_s[a*SDIM + k] + EPSBN);
    WsF[idx] = (_Float16)(W_s[(a*SDIM + k)*HID + n] * sc);
    return;
  }
  idx -= WS_N;
  if (idx < 3*WP_N) {                                  // W_k / W_v / W_sel (head-concat [128,128])
    int which = (int)(idx / WP_N);
    size_t r = idx % WP_N;
    int f = (int)(r/512), L = (int)((r%512)/16), e = (int)(r%16);
    int kt = f>>3, nt = f&7;
    int k = kt*32 + (L>>4)*16 + e, col = nt*16 + (L&15);
    int h = col >> 5, d = col & 31;
    size_t src = ((size_t)h*HID + k)*DHEAD + d;
    if (which == 0)      WkF[r]   = (_Float16)W_k[src];
    else if (which == 1) WvF[r]   = (_Float16)W_v[src];
    else                 WselF[r] = (_Float16)(W_sel[src] * 0.17677669529663687f); // 1/sqrt(32)
    return;
  }
  idx -= 3*WP_N;
  if (idx < WC1_N) {                                   // W_c1
    size_t a = idx / 32768, r = idx % 32768;
    int f = (int)(r/512), L = (int)((r%512)/16), e = (int)(r%16);
    int kt = f>>3, nt = f&7;
    int k = kt*32 + (L>>4)*16 + e, n = nt*16 + (L&15);
    Wc1F[idx] = (_Float16)W_c1[(a*2*HID + k)*HID + n];
  }
}

__global__ void k_pack_bias(const float* __restrict__ W_enc, const float* __restrict__ b_enc,
                            const float* __restrict__ beta_sa, const float* __restrict__ mu_sa,
                            const float* __restrict__ var_sa,
                            const float* __restrict__ W_s, const float* __restrict__ b_s,
                            const float* __restrict__ beta_s, const float* __restrict__ mu_s,
                            const float* __restrict__ var_s,
                            float* __restrict__ benc, float* __restrict__ bsn) {
  int t = blockIdx.x*blockDim.x + threadIdx.x;
  if (t < AGENTS*HID) {
    int a = t >> 7, n = t & 127;
    float acc = b_enc[t];
    for (int k = 0; k < IDIM; ++k) {
      float sc = rsqrtf(var_sa[a*IDIM + k] + EPSBN);
      float sh = beta_sa[a*IDIM + k] - mu_sa[a*IDIM + k]*sc;
      acc += sh * W_enc[((size_t)a*IDIM + k)*HID + n];
    }
    benc[t] = acc;
  } else if (t < 2*AGENTS*HID) {
    int u = t - AGENTS*HID;
    int a = u >> 7, n = u & 127;
    float acc = b_s[u];
    for (int k = 0; k < SDIM; ++k) {
      float sc = rsqrtf(var_s[a*SDIM + k] + EPSBN);
      float sh = beta_s[a*SDIM + k] - mu_s[a*SDIM + k]*sc;
      acc += sh * W_s[((size_t)a*SDIM + k)*HID + n];
    }
    bsn[u] = acc;
  }
}

// -------- fused encoder + projections: e,s_e live only in LDS; keys/vals/sel/s_e -> ws --------
__global__ __launch_bounds__(128) void k_encode(
    const float* __restrict__ states, const float* __restrict__ actions,
    const _Float16* __restrict__ WencF, const _Float16* __restrict__ WsF,
    const _Float16* __restrict__ WkF, const _Float16* __restrict__ WvF,
    const _Float16* __restrict__ WselF,
    const float* __restrict__ benc, const float* __restrict__ bsn,
    _Float16* __restrict__ se_g, _Float16* __restrict__ keys_g,
    _Float16* __restrict__ vals_g, _Float16* __restrict__ sel_g) {
  __shared__ _Float16 Lsa[BT*SA_STR];
  __shared__ _Float16 Le [BT*E_STR];
  __shared__ _Float16 Lse[BT*E_STR];
  const int a = blockIdx.y;
  const size_t rowbase = (size_t)blockIdx.x * BT;
  const int tid = threadIdx.x, lane = tid & 31, wv = tid >> 5;
  const int lh = lane >> 4, lm = lane & 15, wr = wv * 16;

  // stage [states||actions] tile as f16 in LDS (BN folded into weights -> raw inputs)
  const float* sp = states  + ((size_t)a*BATCH + rowbase) * SDIM;
  const float* ap = actions + ((size_t)a*BATCH + rowbase) * ADIM;
  for (int it = tid; it < BT*(IDIM/4); it += 128) {
    int r = it / (IDIM/4), c = (it % (IDIM/4)) * 4;
    float4 x = (c < SDIM) ? *(const float4*)(sp + (size_t)r*SDIM + c)
                          : *(const float4*)(ap + (size_t)r*ADIM + (c - SDIM));
    v4h h; h[0]=(_Float16)x.x; h[1]=(_Float16)x.y; h[2]=(_Float16)x.z; h[3]=(_Float16)x.w;
    *(v4h*)&Lsa[r*SA_STR + c] = h;
  }
  __syncthreads();

  v8f zero = {};
  v8f acc[8];
  // wave-private bounce region for transposing D-frags to row-major (reuses Lsa rows)
  _Float16* bounce = Lsa + wr*SA_STR;

  // e = leaky(sa' @ Wenc' + benc')
  #pragma unroll
  for (int i = 0; i < 8; ++i) acc[i] = zero;
  gemm_tile<9>(Lsa + wr*SA_STR, SA_STR, WencF + (size_t)a*9*8*512, acc, lane);
  #pragma unroll
  for (int nt = 0; nt < 8; ++nt) {
    float bv = benc[a*HID + nt*16 + lm];
    #pragma unroll
    for (int r = 0; r < 8; ++r) {
      float x = acc[nt][r] + bv; x = x > 0.f ? x : ALPHA*x;
      Le[(wr + r + 8*lh)*E_STR + nt*16 + lm] = (_Float16)x;
    }
  }

  // s_e = leaky(s' @ Ws' + bs') ; kept in LDS and persisted (coalesced) for critic
  #pragma unroll
  for (int i = 0; i < 8; ++i) acc[i] = zero;
  gemm_tile<8>(Lsa + wr*SA_STR, SA_STR, WsF + (size_t)a*8*8*512, acc, lane);
  #pragma unroll
  for (int nt = 0; nt < 8; ++nt) {
    float bv = bsn[a*HID + nt*16 + lm];
    #pragma unroll
    for (int r = 0; r < 8; ++r) {
      float x = acc[nt][r] + bv; x = x > 0.f ? x : ALPHA*x;
      Lse[(wr + r + 8*lh)*E_STR + nt*16 + lm] = (_Float16)x;
    }
  }
  // coalesced s_e store straight from Lse (two full 256B rows per b128 instr)
  #pragma unroll
  for (int i = 0; i < 8; ++i) {
    int q = i*32 + lane, row = q >> 4, ch = q & 15;
    v8h x = *(const v8h*)&Lse[(wr + row)*E_STR + ch*8];
    *(v8h*)&se_g[((size_t)a*BATCH + rowbase + wr + row)*HID + ch*8] = x;
  }

  // keys = e @ Wk
  #pragma unroll
  for (int i = 0; i < 8; ++i) acc[i] = zero;
  gemm_tile<4>(Le + wr*E_STR, E_STR, WkF, acc, lane);
  #pragma unroll
  for (int nt = 0; nt < 8; ++nt)
    #pragma unroll
    for (int r = 0; r < 8; ++r)
      bounce[(r + 8*lh)*E_STR + nt*16 + lm] = (_Float16)acc[nt][r];
  #pragma unroll
  for (int i = 0; i < 8; ++i) {
    int q = i*32 + lane, row = q >> 4, ch = q & 15;
    v8h x = *(const v8h*)&bounce[row*E_STR + ch*8];
    *(v8h*)&keys_g[((size_t)a*BATCH + rowbase + wr + row)*HID + ch*8] = x;
  }

  // vals = leaky(e @ Wv)
  #pragma unroll
  for (int i = 0; i < 8; ++i) acc[i] = zero;
  gemm_tile<4>(Le + wr*E_STR, E_STR, WvF, acc, lane);
  #pragma unroll
  for (int nt = 0; nt < 8; ++nt)
    #pragma unroll
    for (int r = 0; r < 8; ++r) {
      float x = acc[nt][r]; x = x > 0.f ? x : ALPHA*x;
      bounce[(r + 8*lh)*E_STR + nt*16 + lm] = (_Float16)x;
    }
  #pragma unroll
  for (int i = 0; i < 8; ++i) {
    int q = i*32 + lane, row = q >> 4, ch = q & 15;
    v8h x = *(const v8h*)&bounce[row*E_STR + ch*8];
    *(v8h*)&vals_g[((size_t)a*BATCH + rowbase + wr + row)*HID + ch*8] = x;
  }

  // sel = s_e @ (Wsel / sqrt(D))
  #pragma unroll
  for (int i = 0; i < 8; ++i) acc[i] = zero;
  gemm_tile<4>(Lse + wr*E_STR, E_STR, WselF, acc, lane);
  #pragma unroll
  for (int nt = 0; nt < 8; ++nt)
    #pragma unroll
    for (int r = 0; r < 8; ++r)
      bounce[(r + 8*lh)*E_STR + nt*16 + lm] = (_Float16)acc[nt][r];
  #pragma unroll
  for (int i = 0; i < 8; ++i) {
    int q = i*32 + lane, row = q >> 4, ch = q & 15;
    v8h x = *(const v8h*)&bounce[row*E_STR + ch*8];
    *(v8h*)&sel_g[((size_t)a*BATCH + rowbase + wr + row)*HID + ch*8] = x;
  }
}

// -------- 8x8 self-excluding softmax attention per (head, batch elem) --------
__global__ __launch_bounds__(256) void k_attn(
    const _Float16* __restrict__ keys_g, const _Float16* __restrict__ vals_g,
    const _Float16* __restrict__ sel_g, _Float16* __restrict__ other_g) {
  __shared__ _Float16 Lk[AGENTS*TB*AT_STR];
  __shared__ _Float16 Lv[AGENTS*TB*AT_STR];
  __shared__ _Float16 Ls[AGENTS*TB*AT_STR];
  const size_t bbase = (size_t)blockIdx.x * TB;
  const int tid = threadIdx.x;

  // async global->LDS staging (ASYNCcnt), 16B per lane, uniform SGPR bases
  for (int it = tid; it < AGENTS*TB*(HID/8); it += 256) {
    int j = it / (TB*16), rem = it % (TB*16);
    int bb = rem / 16, c = (rem % 16) * 8;
    unsigned voff = (unsigned)(((((unsigned)j*BATCH) + (unsigned)bbase + bb)*HID + c) * 2);
    int l = (j*TB + bb)*AT_STR + c;
    async_cp16(lds_off(&Lk[l]), keys_g, voff);
    async_cp16(lds_off(&Lv[l]), vals_g, voff);
    async_cp16(lds_off(&Ls[l]), sel_g, voff);
  }
  wait_async0();
  __syncthreads();

  const int b = tid & 7, i = (tid >> 3) & 7, n = tid >> 6;   // agent i, head n
  float sr[DHEAD];
  {
    const _Float16* p = &Ls[(i*TB + b)*AT_STR + n*DHEAD];
    #pragma unroll
    for (int c = 0; c < 8; ++c) {
      v4h x = *(const v4h*)(p + c*4);
      sr[c*4+0]=(float)x[0]; sr[c*4+1]=(float)x[1]; sr[c*4+2]=(float)x[2]; sr[c*4+3]=(float)x[3];
    }
  }
  float lg[AGENTS], mx = -3.4e38f;
  #pragma unroll
  for (int j = 0; j < AGENTS; ++j) {
    const _Float16* kp = &Lk[(j*TB + b)*AT_STR + n*DHEAD];
    float s = 0.f;
    #pragma unroll
    for (int c = 0; c < 8; ++c) {
      v4h x = *(const v4h*)(kp + c*4);
      s += sr[c*4+0]*(float)x[0] + sr[c*4+1]*(float)x[1]
         + sr[c*4+2]*(float)x[2] + sr[c*4+3]*(float)x[3];
    }
    s = (j == i) ? -1e9f : s;                 // exclude self (1/sqrt(D) already in W_sel)
    lg[j] = s; mx = fmaxf(mx, s);
  }
  float den = 0.f, pr[AGENTS];
  #pragma unroll
  for (int j = 0; j < AGENTS; ++j) { float e = __expf(lg[j] - mx); pr[j] = e; den += e; }
  float inv = 1.f / den;
  float out[DHEAD];
  #pragma unroll
  for (int d = 0; d < DHEAD; ++d) out[d] = 0.f;
  #pragma unroll
  for (int j = 0; j < AGENTS; ++j) {
    float pj = pr[j] * inv;
    const _Float16* vp = &Lv[(j*TB + b)*AT_STR + n*DHEAD];
    #pragma unroll
    for (int c = 0; c < 8; ++c) {
      v4h x = *(const v4h*)(vp + c*4);
      out[c*4+0] += pj*(float)x[0]; out[c*4+1] += pj*(float)x[1];
      out[c*4+2] += pj*(float)x[2]; out[c*4+3] += pj*(float)x[3];
    }
  }
  _Float16* op = other_g + ((size_t)i*BATCH + bbase + b)*HID + n*DHEAD;
  #pragma unroll
  for (int c = 0; c < 4; ++c) {
    v8h o;
    #pragma unroll
    for (int u = 0; u < 8; ++u) o[u] = (_Float16)out[c*8+u];
    *(v8h*)(op + c*8) = o;
  }
}

// -------- critic: h1 = leaky([s_e||other] @ Wc1 + b), q = h1 . w2 + b2 --------
__global__ __launch_bounds__(128) void k_critic(
    const _Float16* __restrict__ se_g, const _Float16* __restrict__ other_g,
    const _Float16* __restrict__ Wc1F, const float* __restrict__ b_c1,
    const float* __restrict__ W_c2, const float* __restrict__ b_c2,
    float* __restrict__ qout) {
  __shared__ _Float16 Lc[BT*C_STR];
  const int a = blockIdx.y;
  const size_t rowbase = (size_t)blockIdx.x * BT;
  const int tid = threadIdx.x, lane = tid & 31, wv = tid >> 5;
  const int lh = lane >> 4, lm = lane & 15, wr = wv * 16;

  // async staging: cols 0..127 from s_e, cols 128..255 from other (uniform bases)
  const unsigned gbase = (unsigned)(((unsigned)a*BATCH + (unsigned)rowbase) * HID);
  for (int it = tid; it < BT*16; it += 128) {
    int r = it >> 4, c = (it & 15) * 8;
    unsigned voff = (gbase + (unsigned)r*HID + c) * 2u;
    async_cp16(lds_off(&Lc[r*C_STR + c]), se_g, voff);
  }
  for (int it = tid; it < BT*16; it += 128) {
    int r = it >> 4, c = (it & 15) * 8;
    unsigned voff = (gbase + (unsigned)r*HID + c) * 2u;
    async_cp16(lds_off(&Lc[r*C_STR + HID + c]), other_g, voff);
  }
  wait_async0();
  __syncthreads();

  v8f zero = {};
  v8f acc[8];
  #pragma unroll
  for (int i = 0; i < 8; ++i) acc[i] = zero;
  gemm_tile<8>(Lc + wr*C_STR, C_STR, Wc1F + (size_t)a*8*8*512, acc, lane);

  float q[8];
  #pragma unroll
  for (int r = 0; r < 8; ++r) q[r] = 0.f;
  #pragma unroll
  for (int nt = 0; nt < 8; ++nt) {
    float bv = b_c1[a*HID + nt*16 + lm];
    float w2 = W_c2[a*HID + nt*16 + lm];      // W_c2 is [A,128,1]
    #pragma unroll
    for (int r = 0; r < 8; ++r) {
      float x = acc[nt][r] + bv; x = x > 0.f ? x : ALPHA*x;
      q[r] += x * w2;
    }
  }
  // reduce the 16 column-lanes within each half-wave
  #pragma unroll
  for (int off = 1; off < 16; off <<= 1)
    #pragma unroll
    for (int r = 0; r < 8; ++r) q[r] += __shfl_xor(q[r], off, 32);
  if (lm == 0) {
    float bq = b_c2[a];
    #pragma unroll
    for (int r = 0; r < 8; ++r)
      qout[(size_t)a*BATCH + rowbase + wr + r + 8*lh] = q[r] + bq;
  }
}

extern "C" void kernel_launch(void* const* d_in, const int* in_sizes, int n_in,
                              void* d_out, int out_size, void* d_ws, size_t ws_size,
                              hipStream_t stream) {
  (void)in_sizes; (void)n_in; (void)out_size; (void)ws_size;
  const float* states  = (const float*)d_in[0];
  const float* actions = (const float*)d_in[1];
  const float* beta_sa = (const float*)d_in[2];
  const float* mu_sa   = (const float*)d_in[3];
  const float* var_sa  = (const float*)d_in[4];
  const float* W_enc   = (const float*)d_in[5];
  const float* b_enc   = (const float*)d_in[6];
  const float* beta_s  = (const float*)d_in[7];
  const float* mu_s    = (const float*)d_in[8];
  const float* var_s   = (const float*)d_in[9];
  const float* W_s     = (const float*)d_in[10];
  const float* b_s     = (const float*)d_in[11];
  const float* W_k     = (const float*)d_in[12];
  const float* W_sel   = (const float*)d_in[13];
  const float* W_v     = (const float*)d_in[14];
  const float* W_c1    = (const float*)d_in[15];
  const float* b_c1    = (const float*)d_in[16];
  const float* W_c2    = (const float*)d_in[17];
  const float* b_c2    = (const float*)d_in[18];
  // d_in[19] action_idx: all-q has a single column -> gather is the identity.

  char* ws = (char*)d_ws;
  size_t off = 0;
  auto take = [&](size_t bytes) -> char* {
    char* p = ws + off; off = (off + bytes + 255) & ~(size_t)255; return p;
  };
  _Float16* WencF = (_Float16*)take(WENC_N * 2);
  _Float16* WsF   = (_Float16*)take(WS_N * 2);
  _Float16* WkF   = (_Float16*)take(WP_N * 2);
  _Float16* WvF   = (_Float16*)take(WP_N * 2);
  _Float16* WselF = (_Float16*)take(WP_N * 2);
  _Float16* Wc1F  = (_Float16*)take(WC1_N * 2);
  float*    benc  = (float*)take(AGENTS*HID * 4);
  float*    bsn   = (float*)take(AGENTS*HID * 4);
  const size_t INTER = (size_t)AGENTS*BATCH*HID*2;
  _Float16* se_g    = (_Float16*)take(INTER);
  _Float16* keys_g  = (_Float16*)take(INTER);
  _Float16* vals_g  = (_Float16*)take(INTER);
  _Float16* sel_g   = (_Float16*)take(INTER);
  _Float16* other_g = (_Float16*)take(INTER);

  const size_t packTotal = WENC_N + WS_N + 3*WP_N + WC1_N;     // 868352
  k_pack_frags<<<dim3((unsigned)((packTotal + 255)/256)), dim3(256), 0, stream>>>(
      W_enc, var_sa, W_s, var_s, W_k, W_v, W_sel, W_c1,
      WencF, WsF, WkF, WvF, WselF, Wc1F);
  k_pack_bias<<<dim3(8), dim3(256), 0, stream>>>(
      W_enc, b_enc, beta_sa, mu_sa, var_sa, W_s, b_s, beta_s, mu_s, var_s, benc, bsn);
  k_encode<<<dim3(BATCH/BT, AGENTS), dim3(128), 0, stream>>>(
      states, actions, WencF, WsF, WkF, WvF, WselF, benc, bsn,
      se_g, keys_g, vals_g, sel_g);
  k_attn<<<dim3(BATCH/TB), dim3(256), 0, stream>>>(keys_g, vals_g, sel_g, other_g);
  k_critic<<<dim3(BATCH/BT, AGENTS), dim3(128), 0, stream>>>(
      se_g, other_g, Wc1F, b_c1, W_c2, b_c2, (float*)d_out);
}
